// Channel_Attention_51556787421488
// MI455X (gfx1250) — matrix operand order repaired and checked by reading, not compiled, not run on hardware
//
#include <hip/hip_runtime.h>

// ---------------------------------------------------------------------------
// MI455X (gfx1250) channel-attention pipeline.
// Roofline: ~86 GFLOP vs ~1.4 GB HBM traffic with f16 intermediates -> near
// memory-bound at 23.3 TB/s, so: f16 storage for intermediates, f16 WMMA with
// f32 accumulate (v_wmma_f32_16x16x32_f16) for all GEMM stages, fp32 VALU for
// dwconv / l2norm / softmax. Async global->LDS staging where no dtype
// conversion is needed (proj GEMM).
// ---------------------------------------------------------------------------

typedef __attribute__((ext_vector_type(16))) _Float16 v16h;
typedef __attribute__((ext_vector_type(8)))  float    v8f;

union Frag16 { v16h h; unsigned u[8]; };

__device__ inline v8f zero8() {
  v8f z;
#pragma unroll
  for (int i = 0; i < 8; ++i) z[i] = 0.f;
  return z;
}

// Async global->LDS (CDNA5): available only on the device pass and when the
// toolchain declares the builtin; falls back to plain LDS stores otherwise.
// Probe-derived signature: (int32x4 AS1* gsrc, int32x4 AS3* ldst, i32, i32).
#if defined(__HIP_DEVICE_COMPILE__)
#if __has_builtin(__builtin_amdgcn_global_load_async_to_lds_b128)
#define HAVE_ASYNC_LDS 1
#endif
#endif
#ifndef HAVE_ASYNC_LDS
#define HAVE_ASYNC_LDS 0
#endif

typedef int v4i_ld __attribute__((vector_size(16)));
typedef __attribute__((address_space(1))) v4i_ld* g128_t;   // global int32x4*
typedef __attribute__((address_space(3))) v4i_ld* l128_t;   // LDS    int32x4*

// Problem constants
#define BATCH 4
#define CIN   192
#define C3    576
#define HEADS 6
#define DHEAD 32
#define PSZ   8
#define WIN   64      // PSZ*PSZ
#define HWDIM 256
#define HW2   65536   // HWDIM*HWDIM

// ---------------------------------------------------------------------------
// Kernel 1/3: 1x1 conv as GEMM  Y[b, m, p] = sum_k W[m,k] * X[b, k, p]
// Block: 256 thr (8 waves) -> 64(M) x 128(N) tile, K chunked by 64 in LDS.
// Wave grid 4(M) x 2(N); each wave: 16M x 64N = 4 accumulators, WMMA f16.
// ---------------------------------------------------------------------------
template <int M, int K, bool IN_F32, bool OUT_F16>
__global__ __launch_bounds__(256) void gemm1x1(const void* __restrict__ Xv,
                                               const float* __restrict__ Wt,
                                               void* __restrict__ Yv) {
  constexpr int MT = 64, NT = 128, KC = 64;
  __shared__ __align__(16) _Float16 Ws[MT * KC];
  __shared__ __align__(16) _Float16 Xs[KC * NT];

  const int b  = blockIdx.z;
  const int m0 = blockIdx.y * MT;
  const int p0 = blockIdx.x * NT;
  const float*    Xf = (const float*)Xv    + (size_t)b * K * HW2;
  const _Float16* Xh = (const _Float16*)Xv + (size_t)b * K * HW2;

  const int tid  = threadIdx.x;
  const int wave = tid >> 5, lane = tid & 31;
  const int wm = wave >> 1, wn = wave & 1;
  const int l = lane & 15, hi = lane >> 4;

  v8f acc[4];
#pragma unroll
  for (int nf = 0; nf < 4; ++nf) acc[nf] = zero8();

  for (int kc = 0; kc < K; kc += KC) {
    __syncthreads();
    // stage weights tile (fp32 -> f16)
    for (int idx = tid; idx < MT * KC; idx += 256) {
      int m = idx >> 6, k = idx & (KC - 1);
      Ws[idx] = (_Float16)Wt[(size_t)(m0 + m) * K + kc + k];
    }
    // stage activations tile
    if (IN_F32) {
      for (int idx = tid; idx < KC * NT; idx += 256) {
        int k = idx >> 7, n = idx & (NT - 1);
        Xs[idx] = (_Float16)Xf[(size_t)(kc + k) * HW2 + p0 + n];
      }
    } else {
#if HAVE_ASYNC_LDS
      // input already f16: pure byte copy -> async DMA into LDS (ASYNCcnt)
      for (int t = tid; t < (KC * NT) / 8; t += 256) {
        int k = t >> 4, n = (t & 15) * 8;           // 8 halfs = 16B per op
        const _Float16* g = &Xh[(size_t)(kc + k) * HW2 + p0 + n];
        _Float16*       s = &Xs[k * NT + n];
        __builtin_amdgcn_global_load_async_to_lds_b128(
            (g128_t)(void*)g, (l128_t)(void*)s, 0, 0);
      }
      asm volatile("s_wait_asynccnt 0" ::: "memory");
#else
      for (int idx = tid; idx < KC * NT; idx += 256) {
        int k = idx >> 7, n = idx & (NT - 1);
        Xs[idx] = Xh[(size_t)(kc + k) * HW2 + p0 + n];
      }
#endif
    }
    // prefetch next K-chunk of activations (emits global_prefetch_b8)
    if (kc + KC < K) {
      const void* pf = IN_F32 ? (const void*)&Xf[(size_t)(kc + KC) * HW2 + p0 + (tid & 127)]
                              : (const void*)&Xh[(size_t)(kc + KC) * HW2 + p0 + (tid & 127)];
      __builtin_prefetch(pf, 0, 0);
    }
    __syncthreads();

#pragma unroll
    for (int ks = 0; ks < KC; ks += 32) {
      // A fragment: 16x32 f16, lane l -> row M, K pairs per VGPR
      Frag16 a;
      const int Mr = wm * 16 + l;
#pragma unroll
      for (int r = 0; r < 8; ++r) {
        int k0 = ks + ((r < 4) ? 0 : 16) + hi * 8 + 2 * (r & 3);
        a.u[r] = *(const unsigned*)&Ws[Mr * KC + k0];
      }
#pragma unroll
      for (int nf = 0; nf < 4; ++nf) {
        // B fragment: 32x16 f16, lane -> N column, VGPR r -> K pair (same order as A)
        Frag16 bb;
        const int nb = wn * 64 + nf * 16;
#pragma unroll
        for (int r = 0; r < 8; ++r) {
          int k0 = ks + ((r < 4) ? 0 : 16) + hi * 8 + 2 * (r & 3);
          bb.h[2 * r]     = Xs[k0 * NT + nb + l];
          bb.h[2 * r + 1] = Xs[(k0 + 1) * NT + nb + l];
        }
        acc[nf] = __builtin_amdgcn_wmma_f32_16x16x32_f16(
            false, a.h, false, bb.h, (short)0, acc[nf], false, false);
      }
    }
  }

  // epilogue: C layout -> VGPR r: (M = r + 8*hi, N = lane&15)
#pragma unroll
  for (int nf = 0; nf < 4; ++nf) {
#pragma unroll
    for (int r = 0; r < 8; ++r) {
      int m = m0 + wm * 16 + r + hi * 8;
      int n = p0 + wn * 64 + nf * 16 + l;
      size_t off = (size_t)b * M * HW2 + (size_t)m * HW2 + n;
      if (OUT_F16) ((_Float16*)Yv)[off] = (_Float16)acc[nf][r];
      else         ((float*)Yv)[off]    = acc[nf][r];
    }
  }
}

// ---------------------------------------------------------------------------
// Kernel 2/3: fused dwconv3x3 + windowed channel attention.
// One wave32 per (window, head); 2 heads per 64-thread block.
// lane = channel-within-head (d = 32 rows). dwconv fused via 10x10 halo.
// k is stored POSITION-MAJOR (transposed); each dots B-fragment register
// is packed from two u16 LDS reads DHEAD apart (p = k0, k0+1).
// ---------------------------------------------------------------------------
__global__ __launch_bounds__(64) void attn_win(const _Float16* __restrict__ qkv,
                                               const float* __restrict__ dw_w,
                                               const float* __restrict__ temp,
                                               _Float16* __restrict__ outp) {
  __shared__ __align__(16) _Float16 q_s[2][DHEAD * WIN];   // [i][p]; reused as out staging
  __shared__ __align__(16) _Float16 k_s[2][WIN * DHEAD];   // TRANSPOSED: [p][j]
  __shared__ __align__(16) _Float16 v_s[2][DHEAD * WIN];   // [j][p]
  __shared__ __align__(16) float    d_s[2][DHEAD * DHEAD];
  __shared__ __align__(16) _Float16 a_s[2][DHEAD * DHEAD];

  const int wslot = threadIdx.x >> 5;
  const int lane  = threadIdx.x & 31;
  const int head  = blockIdx.y * 2 + wslot;
  const int w     = blockIdx.x;            // global window id (incl. batch)
  const int b     = w >> 10;
  const int wi    = w & 1023;
  const int wy0   = (wi >> 5) << 3;
  const int wx0   = (wi & 31) << 3;
  const int l  = lane & 15;
  const int hi = lane >> 4;

  // ---- stage 1: depthwise 3x3 (zero-pad) -> q/k/v tiles in LDS (f16) ------
  for (int sel = 0; sel < 3; ++sel) {
    const int ch = sel * CIN + head * DHEAD + lane;
    const _Float16* bp = qkv + ((size_t)b * C3 + ch) * HW2;
    float dwk[9];
#pragma unroll
    for (int t = 0; t < 9; ++t) dwk[t] = dw_w[ch * 9 + t];
    _Float16* dst = (sel == 0) ? &q_s[wslot][0]
                  : (sel == 1) ? &k_s[wslot][0] : &v_s[wslot][0];
    const bool trans = (sel == 1);

    float r0[10], r1[10], r2[10];
    // row loader with zero padding
    {
      int y = wy0 - 1;
#pragma unroll
      for (int x = 0; x < 10; ++x) {
        int gx = wx0 - 1 + x;
        r0[x] = ((unsigned)y < 256u && (unsigned)gx < 256u) ? (float)bp[y * HWDIM + gx] : 0.f;
      }
      y = wy0;
#pragma unroll
      for (int x = 0; x < 10; ++x) {
        int gx = wx0 - 1 + x;
        r1[x] = ((unsigned)gx < 256u) ? (float)bp[y * HWDIM + gx] : 0.f;
      }
    }
    for (int py = 0; py < 8; ++py) {
      const int y = wy0 + py + 1;
#pragma unroll
      for (int x = 0; x < 10; ++x) {
        int gx = wx0 - 1 + x;
        r2[x] = ((unsigned)y < 256u && (unsigned)gx < 256u) ? (float)bp[y * HWDIM + gx] : 0.f;
      }
#pragma unroll
      for (int px = 0; px < 8; ++px) {
        float acc = 0.f;
#pragma unroll
        for (int kx = 0; kx < 3; ++kx) {
          acc += dwk[kx]     * r0[px + kx];
          acc += dwk[3 + kx] * r1[px + kx];
          acc += dwk[6 + kx] * r2[px + kx];
        }
        if (trans) dst[(py * 8 + px) * DHEAD + lane] = (_Float16)acc;   // k^T
        else       dst[lane * WIN + py * 8 + px]     = (_Float16)acc;
      }
#pragma unroll
      for (int x = 0; x < 10; ++x) { r0[x] = r1[x]; r1[x] = r2[x]; }
    }
  }
  __syncthreads();

  // ---- stage 2: l2-normalize rows of q and k over 64 positions ------------
  {
    _Float16* rowq = &q_s[wslot][lane * WIN];
    float sq = 0.f, sk = 0.f;
#pragma unroll
    for (int t = 0; t < WIN; ++t) { float a = (float)rowq[t]; sq += a * a; }
#pragma unroll
    for (int t = 0; t < WIN; ++t) {
      float a = (float)k_s[wslot][t * DHEAD + lane];  // k^T column = k row
      sk += a * a;
    }
    float iq = 1.f / fmaxf(sqrtf(sq), 1e-12f);
    float ik = 1.f / fmaxf(sqrtf(sk), 1e-12f);
#pragma unroll
    for (int t = 0; t < WIN; ++t) rowq[t] = (_Float16)((float)rowq[t] * iq);
#pragma unroll
    for (int t = 0; t < WIN; ++t) {
      _Float16* kp = &k_s[wslot][t * DHEAD + lane];
      *kp = (_Float16)((float)*kp * ik);
    }
  }
  __syncthreads();

  // ---- stage 3: dots[i][j] = sum_p q[i][p]*k[j][p]  (M=N=32, K=64) --------
  const float tscale = temp[head];
#pragma unroll
  for (int mi = 0; mi < 2; ++mi) {
    v8f c[2]; c[0] = zero8(); c[1] = zero8();
#pragma unroll
    for (int ks = 0; ks < 2; ++ks) {
      Frag16 a;
      const int Mr = mi * 16 + l;
#pragma unroll
      for (int r = 0; r < 8; ++r) {
        int k0 = ks * 32 + ((r < 4) ? 0 : 16) + hi * 8 + 2 * (r & 3);
        a.u[r] = *(const unsigned*)&q_s[wslot][Mr * WIN + k0];
      }
#pragma unroll
      for (int nj = 0; nj < 2; ++nj) {
        // B fragment straight out of k^T: lane -> column j, VGPR r -> p pair (same order as A)
        Frag16 bb;
#pragma unroll
        for (int r = 0; r < 8; ++r) {
          int k0 = ks * 32 + ((r < 4) ? 0 : 16) + hi * 8 + 2 * (r & 3);
          bb.h[2 * r]     = k_s[wslot][k0 * DHEAD + nj * 16 + l];
          bb.h[2 * r + 1] = k_s[wslot][(k0 + 1) * DHEAD + nj * 16 + l];
        }
        c[nj] = __builtin_amdgcn_wmma_f32_16x16x32_f16(
            false, a.h, false, bb.h, (short)0, c[nj], false, false);
      }
    }
#pragma unroll
    for (int nj = 0; nj < 2; ++nj)
#pragma unroll
      for (int r = 0; r < 8; ++r) {
        int m = mi * 16 + r + hi * 8;
        int n = nj * 16 + l;
        d_s[wslot][m * DHEAD + n] = c[nj][r] * tscale;
      }
  }
  __syncthreads();

  // ---- stage 4: row softmax (fp32), write attn as f16 ---------------------
  {
    float ev[32];
    float mx = -3.4e38f;
#pragma unroll
    for (int j = 0; j < 32; ++j) { ev[j] = d_s[wslot][lane * DHEAD + j]; mx = fmaxf(mx, ev[j]); }
    float s = 0.f;
#pragma unroll
    for (int j = 0; j < 32; ++j) { float e = __expf(ev[j] - mx); ev[j] = e; s += e; }
    float inv = 1.f / s;
#pragma unroll
    for (int j = 0; j < 32; ++j) a_s[wslot][lane * DHEAD + j] = (_Float16)(ev[j] * inv);
  }
  __syncthreads();

  // ---- stage 5: out[i][p] = sum_j attn[i][j]*v[j][p]  (M=32,N=64,K=32) ----
  _Float16* omat = &q_s[wslot][0];   // q tile is dead; reuse for output staging
#pragma unroll
  for (int mi = 0; mi < 2; ++mi) {
    Frag16 a;
    const int Mr = mi * 16 + l;
#pragma unroll
    for (int r = 0; r < 8; ++r) {
      int k0 = ((r < 4) ? 0 : 16) + hi * 8 + 2 * (r & 3);
      a.u[r] = *(const unsigned*)&a_s[wslot][Mr * DHEAD + k0];
    }
#pragma unroll
    for (int np = 0; np < 4; ++np) {
      Frag16 bb;
#pragma unroll
      for (int r = 0; r < 8; ++r) {
        int k0 = ((r < 4) ? 0 : 16) + hi * 8 + 2 * (r & 3);
        bb.h[2 * r]     = v_s[wslot][k0 * WIN + np * 16 + l];
        bb.h[2 * r + 1] = v_s[wslot][(k0 + 1) * WIN + np * 16 + l];
      }
      v8f c = zero8();
      c = __builtin_amdgcn_wmma_f32_16x16x32_f16(
          false, a.h, false, bb.h, (short)0, c, false, false);
#pragma unroll
      for (int r = 0; r < 8; ++r) {
        int m = mi * 16 + r + hi * 8;
        int n = np * 16 + l;
        omat[m * WIN + n] = (_Float16)c[r];
      }
    }
  }
  __syncthreads();

  // ---- stage 6: coalesced 16B stores back to [B,192,H,W] f16 --------------
  {
    _Float16* ob = outp + ((size_t)b * CIN + head * DHEAD + lane) * HW2
                 + wy0 * HWDIM + wx0;
#pragma unroll
    for (int ph = 0; ph < 8; ++ph) {
      uint4 vdat = *(const uint4*)&omat[lane * WIN + ph * 8];
      *(uint4*)(ob + ph * HWDIM) = vdat;
    }
  }
}

// ---------------------------------------------------------------------------
extern "C" void kernel_launch(void* const* d_in, const int* in_sizes, int n_in,
                              void* d_out, int out_size, void* d_ws, size_t ws_size,
                              hipStream_t stream) {
  (void)in_sizes; (void)n_in; (void)out_size; (void)ws_size;
  const float* x           = (const float*)d_in[0];  // [4,192,256,256]
  const float* qkv_w       = (const float*)d_in[1];  // [576,192]
  const float* dw_w        = (const float*)d_in[2];  // [576,9]
  const float* proj_w      = (const float*)d_in[3];  // [192,192]
  const float* temperature = (const float*)d_in[4];  // [6]

  // workspace carve-out (f16 intermediates): qkv 302MB + attn-out 101MB
  _Float16* qkv  = (_Float16*)d_ws;
  _Float16* attn = (_Float16*)((char*)d_ws + (size_t)BATCH * C3 * HW2 * sizeof(_Float16));
  float* out = (float*)d_out;

  // 1) qkv = conv1x1(x): M=576, K=192
  gemm1x1<C3, CIN, true, true>
      <<<dim3(HW2 / 128, C3 / 64, BATCH), 256, 0, stream>>>((const void*)x, qkv_w, (void*)qkv);

  // 2) fused dwconv3x3 + windowed channel attention
  attn_win<<<dim3(BATCH * 1024, HEADS / 2, 1), 64, 0, stream>>>(qkv, dw_w, temperature, attn);

  // 3) out = conv1x1(attn): M=192, K=192, fp32 output
  gemm1x1<CIN, CIN, false, false>
      <<<dim3(HW2 / 128, CIN / 64, BATCH), 256, 0, stream>>>((const void*)attn, proj_w, (void*)out);
}
